// SRD_65137474011861
// MI455X (gfx1250) — matrix operand order repaired and checked by reading, not compiled, not run on hardware
//
#include <hip/hip_runtime.h>
#include <hip/hip_bf16.h>
#include <math.h>

// Problem constants (match reference)
#define BQ  4096
#define TT  8
#define HH  512
#define G4H 2048
#define KFULL 512
#define KHALF 256

typedef __attribute__((ext_vector_type(16))) __bf16 bf16x16;
typedef __attribute__((ext_vector_type(8)))  float  f32x8;

#if defined(__has_builtin)
#if __has_builtin(__builtin_amdgcn_tensor_load_to_lds) && __has_builtin(__builtin_amdgcn_s_wait_tensorcnt)
#define HAVE_TDM 1
#endif
#endif

// RNE f32->bf16 for two values, packed into one dword with a single v_perm_b32
__device__ __forceinline__ unsigned int pack2_bf16(float x, float y) {
  unsigned int ux = __float_as_uint(x);
  unsigned int uy = __float_as_uint(y);
  ux += 0x7FFFu + ((ux >> 16) & 1u);   // v_add3_u32
  uy += 0x7FFFu + ((uy >> 16) & 1u);
  // dst bytes {uy[3],uy[2],ux[3],ux[2]} = {bf16(y), bf16(x)}
  return __builtin_amdgcn_perm(uy, ux, 0x07060302u);
}

__device__ __forceinline__ unsigned short f32_to_bf16(float f) {
  unsigned int u = __float_as_uint(f);
  u += 0x7FFFu + ((u >> 16) & 1u);
  return (unsigned short)(u >> 16);
}

#ifdef HAVE_TDM
// ---------------------------------------------------------------------------
// Tensor Data Mover: DMA a 64-row x 256-col bf16 tile (row stride 512 elems in
// memory) into LDS, inserting 16B of padding after every 512B row so the LDS
// row stride is 528B (=> conflict-free ds_load_b128 fragment reads).
// D# per CDNA5 ISA 8.3/8.4: group0 = {count/type/lds/global addr},
// group1 = {data_size, pad, dims, strides}; groups 2/3 zero (2D tensor).
// ---------------------------------------------------------------------------
typedef unsigned int tdm_u32x4 __attribute__((ext_vector_type(4)));
typedef int          tdm_i32x8 __attribute__((ext_vector_type(8)));
typedef int          tdm_i32x4 __attribute__((ext_vector_type(4)));

__device__ __forceinline__ void tdm_load_w_tile(const unsigned short* gsrc,
                                                unsigned int lds_byte_addr) {
  unsigned long long ga = (unsigned long long)(uintptr_t)gsrc;
  tdm_u32x4 g0;
  g0[0] = 1u;                                     // count=1, user mode, no gather
  g0[1] = lds_byte_addr;                          // lds_addr
  g0[2] = (unsigned int)(ga & 0xFFFFFFFFu);       // global_addr[31:0]
  g0[3] = (unsigned int)((ga >> 32) & 0x01FFFFFFu) | (2u << 30);  // addr[56:32] | type=2
  tdm_i32x8 g1;
  g1[0] = (1 << 16)        // data_size = 2 bytes
        | (1 << 20)        // pad_enable
        | (6 << 22)        // pad_interval: 128 dwords = 512B (one tile row)
        | (3 << 25);       // pad_amount:   4 dwords = 16B
  g1[1] = (int)((unsigned)KHALF << 16);           // tensor_dim0 = 256 (low16)
  g1[2] = (int)(64u << 16);                       // tensor_dim0 hi=0 | tensor_dim1=64 (low16)
  g1[3] = (int)((unsigned)KHALF << 16);           // tensor_dim1 hi=0 | tile_dim0 = 256
  g1[4] = 64;                                     // tile_dim1 = 64 | tile_dim2 = 0
  g1[5] = KFULL;                                  // tensor_dim0_stride = 512 elems (low32)
  g1[6] = 0;                                      // stride hi | tensor_dim1_stride lo
  g1[7] = 0;
  tdm_i32x4 z = {0, 0, 0, 0};
#if defined(__clang_major__) && (__clang_major__ >= 23)
  tdm_i32x8 z8 = {0, 0, 0, 0, 0, 0, 0, 0};
  __builtin_amdgcn_tensor_load_to_lds(g0, g1, z, z, z8, 0);
#else
  __builtin_amdgcn_tensor_load_to_lds(g0, g1, z, z, 0);
#endif
}
#endif  // HAVE_TDM

// ---------------------------------------------------------------------------
// f32 -> bf16 weight conversion (done once; weights are step-invariant)
// ---------------------------------------------------------------------------
__global__ void cvt_bf16_kernel(const float* __restrict__ src,
                                unsigned short* __restrict__ dst, int n) {
  int i = blockIdx.x * blockDim.x + threadIdx.x;
  if (i < n) dst[i] = f32_to_bf16(src[i]);
}

// ---------------------------------------------------------------------------
// inp0 = conv-mix over tuple dim; zero h, c, mask
// ---------------------------------------------------------------------------
__global__ void init_kernel(const float* __restrict__ enc,
                            const float* __restrict__ conv_w,
                            const float* __restrict__ conv_b,
                            float* __restrict__ inp, float* __restrict__ h,
                            float* __restrict__ c, int* __restrict__ mask) {
  int tid = blockIdx.x * blockDim.x + threadIdx.x;
  if (tid >= BQ * HH) return;
  int b = tid >> 9;
  int j = tid & (HH - 1);
  float s = 0.f;
#pragma unroll
  for (int t = 0; t < TT; ++t)
    s += enc[((size_t)b * TT + t) * HH + j] * conv_w[t];
  inp[tid] = s + conv_b[0];
  h[tid] = 0.f;
  c[tid] = 0.f;
  if (j < TT) mask[b * TT + j] = 0;
}

// ---------------------------------------------------------------------------
// WMMA GEMM: C[M,N] = A1[M,K=512] * W1^T (+ A2 * W2^T) + bias1 (+ bias2)
//   A* : f32 row-major, converted to bf16 while staging in LDS (float4 loads,
//        double-buffered 64x32 chunks -> stage(i+1) overlaps WMMA(i))
//   W* : bf16 row-major [N,512]; a 64x256 tile is staged into LDS by the TDM
//   C  : f32 row-major
// Block: 128 threads = 4 waves; block tile 64x64; wave tile 16x64.
// All call sites have M%64==0, N%64==0 -> no guards, EXEC stays all-ones
// around WMMA as required.
// ---------------------------------------------------------------------------
__global__ __launch_bounds__(128) void wmma_gemm_kernel(
    const float* __restrict__ A1, const float* __restrict__ A2,
    const unsigned short* __restrict__ Wb1, const unsigned short* __restrict__ Wb2,
    const float* __restrict__ bias1, const float* __restrict__ bias2,
    float* __restrict__ C, int M, int N) {
  __shared__ __align__(16) unsigned short sW[64 * 264];    // 64 x 256 (+8 pad) weight tile
  __shared__ __align__(16) unsigned short sA[2][64 * 40];  // double-buffered 64 x 32 (+8) chunk

  const int tid   = threadIdx.x;
  const int lane  = tid & 31;
  const int wid   = tid >> 5;            // wave 0..3 -> M sub-tile
  const int nbase = blockIdx.x * 64;
  const int mblk  = blockIdx.y * 64;

  f32x8 acc[4];
#pragma unroll
  for (int i = 0; i < 4; ++i) acc[i] = (f32x8){0, 0, 0, 0, 0, 0, 0, 0};

  // ISA 16-bit A layout: lane<16 -> K {0..7,16..23}; lane>=16 -> K {8..15,24..31}
  const int mrow  = wid * 16 + (lane & 15);
  const int kbA   = (lane < 16) ? 0 : 8;
  // ISA 16-bit B layout: lane<16 -> K {0..7,16..23} of col (lane&15); lane>=16 -> K {8..15,24..31}
  const int koffB = (lane < 16) ? 0 : 8;

  // Stage one 64x32 f32 chunk of A (global col base kcol) as bf16 into buf
  auto stage_chunk = [&](unsigned short* buf, const float* A, int kcol) {
#pragma unroll
    for (int j = 0; j < 4; ++j) {
      int q = j * 128 + tid;
      int r = q >> 3, c4 = q & 7;
      float4 v = *(const float4*)(A + (size_t)(mblk + r) * KFULL + kcol + c4 * 4);
      *(uint2*)&buf[r * 40 + c4 * 4] =
          make_uint2(pack2_bf16(v.x, v.y), pack2_bf16(v.z, v.w));
    }
  };

  for (int g = 0; g < 2; ++g) {
    const float* A = (g == 0) ? A1 : A2;
    const unsigned short* W = (g == 0) ? Wb1 : Wb2;
    if (A == nullptr) break;

    for (int kh = 0; kh < KFULL; kh += KHALF) {
      __syncthreads();  // previous weight tile + chunks fully consumed
#ifdef HAVE_TDM
      if (wid == 0)
        tdm_load_w_tile(W + (size_t)nbase * KFULL + kh,
                        (unsigned int)(unsigned long long)(uintptr_t)(const void*)&sW[0]);
      stage_chunk(sA[0], A, kh);          // overlaps the tensor DMA
      if (wid == 0) __builtin_amdgcn_s_wait_tensorcnt(0);
#else
      // Fallback: cooperative uint4 copy of the 64x256 bf16 weight tile
#pragma unroll
      for (int j = 0; j < 16; ++j) {
        int q = j * 128 + tid;
        int r = q >> 5, cc = q & 31;
        *(uint4*)&sW[r * 264 + cc * 8] =
            *(const uint4*)(W + (size_t)(nbase + r) * KFULL + kh + cc * 8);
      }
      stage_chunk(sA[0], A, kh);
#endif
      __syncthreads();  // weight tile + chunk 0 visible to all waves

#pragma unroll 2
      for (int ci = 0; ci < KHALF / 32; ++ci) {
        const int kc = ci * 32;
        // Prefetch next chunk into the idle buffer while computing this one
        if (ci + 1 < KHALF / 32) stage_chunk(sA[(ci + 1) & 1], A, kh + kc + 32);

        const unsigned short* cur = sA[ci & 1];
        union { bf16x16 v; uint4 q[2]; } fa;
        fa.q[0] = *(const uint4*)(&cur[mrow * 40 + kbA]);
        fa.q[1] = *(const uint4*)(&cur[mrow * 40 + kbA + 16]);

#pragma unroll
        for (int i = 0; i < 4; ++i) {
          const int wrow = i * 16 + (lane & 15);
          union { bf16x16 v; uint4 q[2]; } fb;
          fb.q[0] = *(const uint4*)(&sW[wrow * 264 + kc + koffB]);
          fb.q[1] = *(const uint4*)(&sW[wrow * 264 + kc + koffB + 16]);
          acc[i] = __builtin_amdgcn_wmma_f32_16x16x32_bf16(
              false, fa.v, false, fb.v, (short)0, acc[i], false, false);
        }
        __syncthreads();  // publish staged chunk; reads of 'cur' already complete
      }
    }
  }

  // Epilogue per ISA C/D layout: lane<16 -> rows 0..7, lane>=16 -> rows 8..15
  const int rowsel = (lane < 16) ? 0 : 8;
#pragma unroll
  for (int i = 0; i < 4; ++i) {
    const int col = nbase + i * 16 + (lane & 15);
    const float bsum = (bias1 ? bias1[col] : 0.f) + (bias2 ? bias2[col] : 0.f);
#pragma unroll
    for (int r = 0; r < 8; ++r) {
      const int row = mblk + wid * 16 + rowsel + r;
      C[(size_t)row * N + col] = acc[i][r] + bsum;
    }
  }
}

// ---------------------------------------------------------------------------
// LSTM cell nonlinearity (torch gate order i,f,g,o)
// ---------------------------------------------------------------------------
__global__ void lstm_cell_kernel(const float* __restrict__ gates,
                                 float* __restrict__ h, float* __restrict__ c) {
  int tid = blockIdx.x * blockDim.x + threadIdx.x;
  if (tid >= BQ * HH) return;
  int b = tid >> 9;
  int j = tid & (HH - 1);
  const float* g = gates + (size_t)b * G4H;
  float ig = g[j], fg = g[HH + j], gg = g[2 * HH + j], og = g[3 * HH + j];
  float si = 1.f / (1.f + __expf(-ig));
  float sf = 1.f / (1.f + __expf(-fg));
  float so = 1.f / (1.f + __expf(-og));
  float cn = sf * c[tid] + si * tanhf(gg);
  c[tid] = cn;
  h[tid] = so * tanhf(cn);
}

// ---------------------------------------------------------------------------
// Attention scores + masked argmax + gather of next decoder input.
// One wave (32 lanes) per batch row. softmax is monotonic, so
// "softmax -> mask=-inf -> argmax" == masked argmax over raw logits; the
// probs output is the RAW Ui (pre-softmax), matching the reference.
// ---------------------------------------------------------------------------
__global__ __launch_bounds__(256) void attn_step_kernel(
    const float* __restrict__ W1E, const float* __restrict__ hW2,
    const float* __restrict__ vw, const float* __restrict__ vb,
    const float* __restrict__ inps, int* __restrict__ mask,
    float* __restrict__ inp_next, float* __restrict__ probs,
    float* __restrict__ ptrs, int step) {
  const int lane = threadIdx.x & 31;
  const int b = blockIdx.x * 8 + (threadIdx.x >> 5);

  const float* w1e = W1E + (size_t)b * (TT * HH);
  const float* hv  = hW2 + (size_t)b * HH;

  float acc[TT];
#pragma unroll
  for (int t = 0; t < TT; ++t) acc[t] = 0.f;

  for (int hh = lane; hh < HH; hh += 32) {
    float v = vw[hh];
    float hc = hv[hh];
#pragma unroll
    for (int t = 0; t < TT; ++t)
      acc[t] += v * tanhf(w1e[t * HH + hh] + hc);
  }

  // wave32 butterfly reduce; every lane ends with the full sum
#pragma unroll
  for (int t = 0; t < TT; ++t) {
#pragma unroll
    for (int off = 16; off > 0; off >>= 1)
      acc[t] += __shfl_xor(acc[t], off, 32);
    acc[t] += vb[0];
  }

  int mk[TT];
#pragma unroll
  for (int t = 0; t < TT; ++t) mk[t] = mask[b * TT + t];
  int ptr = 0;
  float best = -INFINITY;
#pragma unroll
  for (int t = 0; t < TT; ++t)
    if (!mk[t] && acc[t] > best) { best = acc[t]; ptr = t; }  // first-max ties

  if (lane < TT) probs[((size_t)b * TT + step) * TT + lane] = acc[lane];
  if (lane == 0) {
    ptrs[(size_t)b * TT + step] = (float)ptr;
    mask[b * TT + ptr] = 1;
  }

  // gather: inp_next[b,:] = inps[b, ptr, :]  (512 floats, float4 per lane)
  const float4* src = (const float4*)(inps + ((size_t)b * TT + ptr) * HH);
  float4* dst = (float4*)(inp_next + (size_t)b * HH);
#pragma unroll
  for (int j = 0; j < 4; ++j) dst[lane + 32 * j] = src[lane + 32 * j];
}

// ---------------------------------------------------------------------------
// Launch
// ---------------------------------------------------------------------------
extern "C" void kernel_launch(void* const* d_in, const int* in_sizes, int n_in,
                              void* d_out, int out_size, void* d_ws, size_t ws_size,
                              hipStream_t stream) {
  (void)in_sizes; (void)n_in; (void)out_size; (void)ws_size;
  const float* inps   = (const float*)d_in[0];
  const float* enc    = (const float*)d_in[1];
  const float* conv_w = (const float*)d_in[2];
  const float* conv_b = (const float*)d_in[3];
  const float* Wih    = (const float*)d_in[4];
  const float* Whh    = (const float*)d_in[5];
  const float* bih    = (const float*)d_in[6];
  const float* bhh    = (const float*)d_in[7];
  const float* W1     = (const float*)d_in[8];
  const float* b1     = (const float*)d_in[9];
  const float* W2     = (const float*)d_in[10];
  const float* b2     = (const float*)d_in[11];
  const float* vw     = (const float*)d_in[12];
  const float* vb     = (const float*)d_in[13];

  char* ws = (char*)d_ws;
  size_t off = 0;
  auto alloc = [&](size_t bytes) {
    off = (off + 255) & ~(size_t)255;
    char* p = ws + off;
    off += bytes;
    return p;
  };
  unsigned short* Wih_b = (unsigned short*)alloc((size_t)G4H * HH * 2);
  unsigned short* Whh_b = (unsigned short*)alloc((size_t)G4H * HH * 2);
  unsigned short* W1_b  = (unsigned short*)alloc((size_t)HH * HH * 2);
  unsigned short* W2_b  = (unsigned short*)alloc((size_t)HH * HH * 2);
  float* W1E   = (float*)alloc((size_t)BQ * TT * HH * 4);
  float* gates = (float*)alloc((size_t)BQ * G4H * 4);
  float* hbuf  = (float*)alloc((size_t)BQ * HH * 4);
  float* cbuf  = (float*)alloc((size_t)BQ * HH * 4);
  float* ibuf  = (float*)alloc((size_t)BQ * HH * 4);
  float* hw2   = (float*)alloc((size_t)BQ * HH * 4);
  int*   mask  = (int*)alloc((size_t)BQ * TT * 4);

  float* probs = (float*)d_out;                         // [B,T,T]
  float* ptrs  = (float*)d_out + (size_t)BQ * TT * TT;  // [B,T] (indices as f32)

  // One-time prep
  cvt_bf16_kernel<<<(G4H * HH + 255) / 256, 256, 0, stream>>>(Wih, Wih_b, G4H * HH);
  cvt_bf16_kernel<<<(G4H * HH + 255) / 256, 256, 0, stream>>>(Whh, Whh_b, G4H * HH);
  cvt_bf16_kernel<<<(HH * HH + 255) / 256, 256, 0, stream>>>(W1, W1_b, HH * HH);
  cvt_bf16_kernel<<<(HH * HH + 255) / 256, 256, 0, stream>>>(W2, W2_b, HH * HH);
  init_kernel<<<(BQ * HH) / 256, 256, 0, stream>>>(enc, conv_w, conv_b, ibuf, hbuf, cbuf, mask);
  // W1E = enc @ W1^T + b1   (step-invariant, hoisted)
  wmma_gemm_kernel<<<dim3(HH / 64, (BQ * TT) / 64), 128, 0, stream>>>(
      enc, nullptr, W1_b, nullptr, b1, nullptr, W1E, BQ * TT, HH);

  // Serial decode over T steps
  for (int step = 0; step < TT; ++step) {
    // gates = inp @ Wih^T + h @ Whh^T + bih + bhh
    wmma_gemm_kernel<<<dim3(G4H / 64, BQ / 64), 128, 0, stream>>>(
        ibuf, hbuf, Wih_b, Whh_b, bih, bhh, gates, BQ, G4H);
    lstm_cell_kernel<<<(BQ * HH) / 256, 256, 0, stream>>>(gates, hbuf, cbuf);
    // hW2 = h @ W2^T + b2
    wmma_gemm_kernel<<<dim3(HH / 64, BQ / 64), 128, 0, stream>>>(
        hbuf, nullptr, W2_b, nullptr, b2, nullptr, hw2, BQ, HH);
    attn_step_kernel<<<BQ / 8, 256, 0, stream>>>(
        W1E, hw2, vw, vb, inps, mask, ibuf, probs, ptrs, step);
  }
}
